// TFLSHAttention_79903571575026
// MI455X (gfx1250) — matrix operand order repaired and checked by reading, not compiled, not run on hardware
//
#include <hip/hip_runtime.h>
#include <math.h>

#define B_    16
#define S_    4096
#define D_    64
#define H_    8
#define NBK   64            // buckets per hash round
#define NB_TOT 512          // H_ * NBK
#define CH    512           // chunks per batch
#define HS    (H_ * S_)     // 32768
#define NEGV  (-5.0e4f)

typedef __attribute__((ext_vector_type(16))) _Float16 v16h;
typedef __attribute__((ext_vector_type(8)))  _Float16 v8h;
typedef __attribute__((ext_vector_type(8)))  float    v8f;
typedef __attribute__((ext_vector_type(4)))  unsigned int u32x4;
typedef __attribute__((ext_vector_type(8)))  int i32x8;
typedef __attribute__((ext_vector_type(4)))  int i32x4;

union V16U { v16h v; v8h h[2]; };

// A-operand (16xK tile, 16-bit): lane grp0 e0..7 -> K0..7, e8..15 -> K16..23;
// grp1 shifted by 8.  Two 16B LDS loads.
__device__ __forceinline__ v16h frag_a(const _Float16* row, int kBase, int grp) {
  V16U u;
  const int o = kBase + grp * 8;
  u.h[0] = *(const v8h*)(row + o);
  u.h[1] = *(const v8h*)(row + o + 16);
  return u.v;
}
// B-operand (Kx16 tile, 16-bit): lane grp0 e0..7 -> K0..7, e8..15 -> K16..23; grp1 shifted by 8.
__device__ __forceinline__ v16h frag_b(const _Float16* row, int kBase, int grp) {
  V16U u;
  const int o = kBase + grp * 8;
  u.h[0] = *(const v8h*)(row + o);
  u.h[1] = *(const v8h*)(row + o + 16);
  return u.v;
}
#define WMMA_F32(a, b, c) \
  __builtin_amdgcn_wmma_f32_16x16x32_f16(false, (a), false, (b), (short)0, (c), false, false)

// ---------------- zero output ----------------
__global__ void k_zero(float* __restrict__ out, int n) {
  int i = blockIdx.x * blockDim.x + threadIdx.x;
  if (i < n) out[i] = 0.f;
}

// ---------------- LSH hashing: buckets = argmax([rot, -rot]) + h*64 ----------------
// Rotation tile for fixed (b,h): 64 rows x 32 f32, row stride 256 floats -> one
// TDM descriptor load into LDS (packed), issued by wave 0 only.
__global__ __launch_bounds__(256) void k_hash(const float* __restrict__ qk,
                                              const float* __restrict__ rot,
                                              int* __restrict__ bkt) {
  __shared__ __align__(16) float R[64 * 32];
  const int bh = blockIdx.x >> 4;
  const int b = bh >> 3, h = bh & 7;
  const int t = ((blockIdx.x & 15) << 8) + threadIdx.x;
#if __has_builtin(__builtin_amdgcn_tensor_load_to_lds) && __has_builtin(__builtin_amdgcn_s_wait_tensorcnt)
  if (threadIdx.x == 0) {
    const float* gp = rot + ((size_t)(b * 512 + h)) * 32;   // rot[b][0][h][0]
    const unsigned long long ga = (unsigned long long)(uintptr_t)gp;
    const unsigned ldsA = (unsigned)(uintptr_t)(&R[0]);
    // D# group 0: count=1 | lds_addr | global_addr[56:0] | type=2 ("image")
    u32x4 g0;
    g0[0] = 1u;
    g0[1] = ldsA;
    g0[2] = (unsigned)ga;
    g0[3] = ((unsigned)(ga >> 32) & 0x01FFFFFFu) | (2u << 30);
    // D# group 1: data_size=4B; tensor 32x64 (dim0=32, dim1=64), dim0_stride=256;
    // tile 32x64 packed into LDS.
    i32x8 g1;
    g1[0] = 0x00020000;            // workgroup_mask=0, data_size=2 (4 bytes)
    g1[1] = (int)(32u << 16);      // tensor_dim0 = 32
    g1[2] = (int)(64u << 16);      // tensor_dim0 hi=0 | tensor_dim1 = 64
    g1[3] = (int)(32u << 16);      // tensor_dim1 hi=0 | tile_dim0 = 32
    g1[4] = 64;                    // tile_dim1 = 64, tile_dim2 = 0
    g1[5] = 256;                   // tensor_dim0_stride = 256 (low 32)
    g1[6] = 0;                     // stride hi | tensor_dim1_stride lo
    g1[7] = 0;
    i32x4 gz = {};
    i32x8 z8 = {};
    __builtin_amdgcn_tensor_load_to_lds(g0, g1, gz, gz, z8, 0);
    __builtin_amdgcn_s_wait_tensorcnt(0);
  }
#else
  for (int idx = threadIdx.x; idx < 2048; idx += 256) {
    const int f = idx >> 5, i = idx & 31;
    R[idx] = rot[(((size_t)b * 64 + f) * 8 + h) * 32 + i];
  }
#endif
  __syncthreads();
  float q[64];
  const float4* qv = (const float4*)(qk + ((size_t)b * S_ + t) * D_);
#pragma unroll
  for (int i = 0; i < 16; ++i) {
    float4 x = qv[i];
    q[4 * i] = x.x; q[4 * i + 1] = x.y; q[4 * i + 2] = x.z; q[4 * i + 3] = x.w;
  }
  float dv[32];
#pragma unroll 4
  for (int i = 0; i < 32; ++i) {
    float a = 0.f;
#pragma unroll
    for (int f = 0; f < 64; ++f) a = fmaf(q[f], R[f * 32 + i], a);
    dv[i] = a;
  }
  float best = -3.0e38f; int bi = 0;
#pragma unroll
  for (int i = 0; i < 32; ++i) if (dv[i] > best) { best = dv[i]; bi = i; }
#pragma unroll
  for (int i = 0; i < 32; ++i) if (-dv[i] > best) { best = -dv[i]; bi = 32 + i; }
  bkt[(size_t)b * HS + h * S_ + t] = bi + h * NBK;
}

// ---------------- per-batch histogram + exclusive scan ----------------
__global__ void k_hist(const int* __restrict__ bkt, int* __restrict__ offs) {
  __shared__ int hist[NB_TOT];
  const int b = blockIdx.x;
  for (int i = threadIdx.x; i < NB_TOT; i += blockDim.x) hist[i] = 0;
  __syncthreads();
  for (int i = threadIdx.x; i < HS; i += blockDim.x)
    atomicAdd(&hist[bkt[(size_t)b * HS + i]], 1);
  __syncthreads();
  if (threadIdx.x == 0) {
    int run = 0;
    for (int i = 0; i < NB_TOT; ++i) { offs[b * NB_TOT + i] = run; run += hist[i]; }
  }
}

// ---------------- stable counting-sort scatter: one wave per (b,bucket) ----------------
__global__ void k_gather(const int* __restrict__ bkt, const int* __restrict__ offs,
                         int* __restrict__ sidx) {
  const int lane = threadIdx.x & 31;
  const int wv = blockIdx.x * (blockDim.x >> 5) + (threadIdx.x >> 5);
  const int b = wv >> 9, bucket = wv & 511;
  const int h = bucket >> 6;
  const int base = b * HS + h * S_;
  int off = offs[b * NB_TOT + bucket];
  for (int t0 = 0; t0 < S_; t0 += 32) {
    const int t = t0 + lane;
    const int m = (bkt[base + t] == bucket) ? 1 : 0;
    const unsigned mask = (unsigned)__ballot(m);
    if (m) sidx[(size_t)b * HS + off + __popc(mask & ((1u << lane) - 1))] = h * S_ + t;
    off += __popc(mask);
  }
}

// ---------------- pass 1: per-row logsumexp of masked scaled dots ----------------
__global__ __launch_bounds__(128) void k_lse(const float* __restrict__ qk,
                                             const int* __restrict__ sidx,
                                             float* __restrict__ logits) {
  __shared__ __align__(16) _Float16 Aq[64 * 64];    // raw queries
  __shared__ __align__(16) _Float16 Kl[128 * 64];   // unit-norm keys
  __shared__ int tArr[128];
  __shared__ int jArr[64];
  const int b = blockIdx.y, c = blockIdx.x;
  const int tid = threadIdx.x;
  {
    const int pos = (tid < 64) ? (c * 64 + tid)
                               : (((c + CH - 1) & (CH - 1)) * 64 + (tid - 64));
    const int j = sidx[(size_t)b * HS + pos];
    const int tok = j & (S_ - 1);
    tArr[tid] = tok;
    if (tid < 64) jArr[tid] = j;
    float vr[64];
    const float4* src = (const float4*)(qk + ((size_t)b * S_ + tok) * D_);
#pragma unroll
    for (int i = 0; i < 16; ++i) {
      float4 x = src[i];
      vr[4 * i] = x.x; vr[4 * i + 1] = x.y; vr[4 * i + 2] = x.z; vr[4 * i + 3] = x.w;
    }
    float nrm = 0.f;
#pragma unroll
    for (int d = 0; d < 64; ++d) nrm = fmaf(vr[d], vr[d], nrm);
    const float sc = 1.f / (sqrtf(nrm) + 1e-6f);
#pragma unroll
    for (int d = 0; d < 64; ++d) Kl[tid * 64 + d] = (_Float16)(vr[d] * sc);
    if (tid < 64) {
#pragma unroll
      for (int d = 0; d < 64; ++d) Aq[tid * 64 + d] = (_Float16)vr[d];
    }
  }
  __syncthreads();
  const int lane = tid & 31, wv = tid >> 5;
  const int grp = lane >> 4, n16 = lane & 15;
  const int rowA = wv * 16 + n16;
  const v16h a0 = frag_a(Aq + rowA * 64, 0, grp);
  const v16h a1 = frag_a(Aq + rowA * 64, 32, grp);
  v8f acc[8];
#pragma unroll
  for (int tc = 0; tc < 8; ++tc) {
    v8f c8 = {};
    c8 = WMMA_F32(a0, frag_b(Kl + (tc * 16 + n16) * 64, 0, grp), c8);
    c8 = WMMA_F32(a1, frag_b(Kl + (tc * 16 + n16) * 64, 32, grp), c8);
    acc[tc] = c8;
  }
  int tkc[8];
#pragma unroll
  for (int tc = 0; tc < 8; ++tc) tkc[tc] = tArr[tc * 16 + n16];
  float mx[8], lse[8];
#pragma unroll
  for (int r = 0; r < 8; ++r) {
    const int m = wv * 16 + r + 8 * grp;
    const int tq = tArr[m];
    float best = -3.0e38f;
#pragma unroll
    for (int tc = 0; tc < 8; ++tc) {
      const float val = (tq == tkc[tc]) ? NEGV : acc[tc][r] * 0.125f;
      acc[tc][r] = val;
      best = fmaxf(best, val);
    }
    mx[r] = best;
  }
#pragma unroll
  for (int r = 0; r < 8; ++r)
    for (int d = 1; d < 16; d <<= 1) mx[r] = fmaxf(mx[r], __shfl_xor(mx[r], d, 32));
#pragma unroll
  for (int r = 0; r < 8; ++r) {
    float s = 0.f;
#pragma unroll
    for (int tc = 0; tc < 8; ++tc) s += __expf(acc[tc][r] - mx[r]);
    for (int d = 1; d < 16; d <<= 1) s += __shfl_xor(s, d, 32);
    lse[r] = __logf(s) + mx[r];
  }
  if (n16 == 0) {
#pragma unroll
    for (int r = 0; r < 8; ++r) {
      const int m = wv * 16 + r + 8 * grp;
      logits[(size_t)b * HS + jArr[m]] = lse[r];
    }
  }
}

// ---------------- combine weights across hash rounds ----------------
__global__ void k_weight(const float* __restrict__ logits, float* __restrict__ wgt) {
  const int idx = blockIdx.x * blockDim.x + threadIdx.x;
  const int b = idx >> 12, t = idx & (S_ - 1);
  float l[H_], mx = -3.0e38f;
#pragma unroll
  for (int h = 0; h < H_; ++h) {
    l[h] = logits[(size_t)b * HS + h * S_ + t];
    mx = fmaxf(mx, l[h]);
  }
  float s = 0.f;
#pragma unroll
  for (int h = 0; h < H_; ++h) s += __expf(l[h] - mx);
  const float lseT = __logf(s) + mx;
#pragma unroll
  for (int h = 0; h < H_; ++h) wgt[(size_t)b * HS + h * S_ + t] = __expf(l[h] - lseT);
}

// ---------------- pass 2: recompute dots, weighted probs @ V, atomic accumulate ----------------
__global__ __launch_bounds__(128) void k_out(const float* __restrict__ qk,
                                             const float* __restrict__ vin,
                                             const int* __restrict__ sidx,
                                             const float* __restrict__ logits,
                                             const float* __restrict__ wgt,
                                             float* __restrict__ out) {
  __shared__ __align__(16) _Float16 Aq[64 * 64];
  __shared__ __align__(16) _Float16 Kl[128 * 64];
  __shared__ __align__(16) _Float16 Vt[64 * 128];  // [dim][key]
  __shared__ __align__(16) _Float16 P[64 * 128];   // weighted probs, [row][key]
  __shared__ int tArr[128];
  __shared__ float lseA[64], alA[64];
  const int b = blockIdx.y, c = blockIdx.x;
  const int tid = threadIdx.x;
  {
    const int pos = (tid < 64) ? (c * 64 + tid)
                               : (((c + CH - 1) & (CH - 1)) * 64 + (tid - 64));
    const int j = sidx[(size_t)b * HS + pos];
    const int tok = j & (S_ - 1);
    tArr[tid] = tok;
    float vr[64];
    const float4* src = (const float4*)(qk + ((size_t)b * S_ + tok) * D_);
#pragma unroll
    for (int i = 0; i < 16; ++i) {
      float4 x = src[i];
      vr[4 * i] = x.x; vr[4 * i + 1] = x.y; vr[4 * i + 2] = x.z; vr[4 * i + 3] = x.w;
    }
    float nrm = 0.f;
#pragma unroll
    for (int d = 0; d < 64; ++d) nrm = fmaf(vr[d], vr[d], nrm);
    const float sc = 1.f / (sqrtf(nrm) + 1e-6f);
#pragma unroll
    for (int d = 0; d < 64; ++d) Kl[tid * 64 + d] = (_Float16)(vr[d] * sc);
    if (tid < 64) {
#pragma unroll
      for (int d = 0; d < 64; ++d) Aq[tid * 64 + d] = (_Float16)vr[d];
      lseA[tid] = logits[(size_t)b * HS + j];
      alA[tid] = wgt[(size_t)b * HS + j];
    }
    const float4* vsrc = (const float4*)(vin + ((size_t)b * S_ + tok) * D_);
#pragma unroll
    for (int i = 0; i < 16; ++i) {
      float4 x = vsrc[i];
      Vt[(4 * i) * 128 + tid]     = (_Float16)x.x;
      Vt[(4 * i + 1) * 128 + tid] = (_Float16)x.y;
      Vt[(4 * i + 2) * 128 + tid] = (_Float16)x.z;
      Vt[(4 * i + 3) * 128 + tid] = (_Float16)x.w;
    }
  }
  __syncthreads();
  const int lane = tid & 31, wv = tid >> 5;
  const int grp = lane >> 4, n16 = lane & 15;
  const int rowA = wv * 16 + n16;
  const v16h a0 = frag_a(Aq + rowA * 64, 0, grp);
  const v16h a1 = frag_a(Aq + rowA * 64, 32, grp);
  v8f acc[8];
#pragma unroll
  for (int tc = 0; tc < 8; ++tc) {
    v8f c8 = {};
    c8 = WMMA_F32(a0, frag_b(Kl + (tc * 16 + n16) * 64, 0, grp), c8);
    c8 = WMMA_F32(a1, frag_b(Kl + (tc * 16 + n16) * 64, 32, grp), c8);
    acc[tc] = c8;
  }
  int tkc[8];
#pragma unroll
  for (int tc = 0; tc < 8; ++tc) tkc[tc] = tArr[tc * 16 + n16];
#pragma unroll
  for (int r = 0; r < 8; ++r) {
    const int m = wv * 16 + r + 8 * grp;
    const int tq = tArr[m];
    const float lse = lseA[m];
    const float al = alA[m];
#pragma unroll
    for (int tc = 0; tc < 8; ++tc) {
      const float val = (tq == tkc[tc]) ? NEGV : acc[tc][r] * 0.125f;
      P[m * 128 + tc * 16 + n16] = (_Float16)(__expf(val - lse) * al);
    }
  }
  __syncthreads();
  v16h ap[4];
#pragma unroll
  for (int kb = 0; kb < 4; ++kb) ap[kb] = frag_a(P + rowA * 128, kb * 32, grp);
#pragma unroll
  for (int tn = 0; tn < 4; ++tn) {
    v8f c8 = {};
#pragma unroll
    for (int kb = 0; kb < 4; ++kb)
      c8 = WMMA_F32(ap[kb], frag_b(Vt + (tn * 16 + n16) * 128, kb * 32, grp), c8);
#pragma unroll
    for (int r = 0; r < 8; ++r) {
      const int m = wv * 16 + r + 8 * grp;
      const int tok = tArr[m];
      atomicAdd(&out[((size_t)b * S_ + tok) * D_ + tn * 16 + n16], c8[r]);
    }
  }
}

extern "C" void kernel_launch(void* const* d_in, const int* in_sizes, int n_in,
                              void* d_out, int out_size, void* d_ws, size_t ws_size,
                              hipStream_t stream) {
  (void)in_sizes; (void)n_in; (void)out_size; (void)ws_size;
  const float* qk  = (const float*)d_in[0];
  const float* v   = (const float*)d_in[1];
  const float* rot = (const float*)d_in[2];
  float* out = (float*)d_out;
  int* bkt = (int*)((char*)d_out + (size_t)B_ * S_ * D_ * sizeof(float));

  int* offs = (int*)d_ws;                       // B*512 ints
  int* sidx = offs + B_ * NB_TOT;               // B*32768 ints
  float* logits = (float*)(sidx + B_ * HS);     // B*32768 floats
  float* wgt = logits + B_ * HS;                // B*32768 floats

  k_zero<<<(B_ * S_ * D_ + 255) / 256, 256, 0, stream>>>(out, B_ * S_ * D_);
  k_hash<<<B_ * H_ * 16, 256, 0, stream>>>(qk, rot, bkt);
  k_hist<<<B_, 256, 0, stream>>>(bkt, offs);
  k_gather<<<(B_ * NB_TOT) / 8, 256, 0, stream>>>(bkt, offs, sidx);
  k_lse<<<dim3(CH, B_), 128, 0, stream>>>(qk, sidx, logits);
  k_weight<<<(B_ * S_) / 256, 256, 0, stream>>>(logits, wgt);
  k_out<<<dim3(CH, B_), 128, 0, stream>>>(qk, v, sidx, logits, wgt, out);
}
